// SelfAttention_10660108828899
// MI455X (gfx1250) — compile-verified, matrix operand order checked by reading, not run on hardware
//
#include <hip/hip_runtime.h>

// ---------------------------------------------------------------------------
// Self-attention for MI455X (gfx1250, wave32, WMMA bf16).
//   B=4, T=4096, D=1024.
//   Pass 1: Q/K/V projections as bf16 WMMA GEMMs (fp32 accumulate), bf16
//           outputs in workspace. Q pre-scaled by 1/sqrt(D). V written
//           transposed [B][D][T] so PV B-fragments are contiguous loads.
//           Block tile 128x128, 8 waves x (32x64) -> 8 WMMAs per k-step.
//   Pass 2: flash attention, 16 query rows per block, 8 waves split the
//           1024-wide dims; online softmax in fp32; WMMA for QK^T and PV.
// Workspace: 3 * B*T*D * 2 bytes = 96 MB bf16.
// ---------------------------------------------------------------------------

typedef unsigned short u16x8  __attribute__((ext_vector_type(8)));
typedef unsigned short u16x16 __attribute__((ext_vector_type(16)));
typedef __bf16         v16bf  __attribute__((ext_vector_type(16)));
typedef float          v8f    __attribute__((ext_vector_type(8)));
typedef float          f32x4  __attribute__((ext_vector_type(4)));
typedef int            i32x4  __attribute__((ext_vector_type(4)));

#define DMODEL 1024
#define SEQ    4096
#define NBATCH 4

// round-to-nearest-even fp32 -> bf16 (bit pattern as ushort)
__device__ __forceinline__ unsigned short f2bf(float f) {
  unsigned int u = __float_as_uint(f);
  u += 0x7FFFu + ((u >> 16) & 1u);
  return (unsigned short)(u >> 16);
}

// Build a 16x32 (or 32x16) bf16 WMMA fragment for this lane.
// p points at the lane's K-offset base ( (lane>>4)*8 ), 16B aligned.
// K(j) = base + (j&7) + 16*(j>>3)  ->  two contiguous 8-half (16B) loads.
__device__ __forceinline__ v16bf load_frag16(const unsigned short* p) {
  u16x8 lo = *(const u16x8*)p;
  u16x8 hi = *(const u16x8*)(p + 16);
  u16x16 full = __builtin_shufflevector(lo, hi,
      0, 1, 2, 3, 4, 5, 6, 7, 8, 9, 10, 11, 12, 13, 14, 15);
  return __builtin_bit_cast(v16bf, full);
}

__device__ __forceinline__ v8f wmma_bf16(v16bf a, v16bf b, v8f c) {
  // D(16x16 f32) = A(16x32 bf16) * B(32x16 bf16) + C
  return __builtin_amdgcn_wmma_f32_16x16x32_bf16(
      false, a, false, b, (short)0, c, false, false);
}

// ---------------------------------------------------------------------------
// Pass 1: C[M=B*T, N=D] = X[M,D] * W[D,N] + bias, output bf16 (optionally
// scaled, optionally transposed to [B][D][T]).
// Block tile 128x128, 8 waves in 4x2; each wave a 32x64 tile (8 accums).
// ---------------------------------------------------------------------------
__global__ __launch_bounds__(256)
void proj_gemm_bf16(const float* __restrict__ X, const float* __restrict__ W,
                    const float* __restrict__ bias,
                    unsigned short* __restrict__ Out,
                    float scale, int transpose_out) {
  __shared__ __align__(16) unsigned short As[128][40]; // 128x32 bf16 (+pad)
  __shared__ __align__(16) unsigned short Bt[128][40]; // W^T tile: [n][k]

  const int tid  = threadIdx.x;
  const int wave = tid >> 5;
  const int lane = tid & 31;
  const int l15  = lane & 15;
  const int lhi  = (lane >> 4) * 8;
  const int wr   = wave >> 1;       // 0..3 : 32-row group
  const int wc   = wave & 1;        // 0..1 : 64-col group
  const int m0   = blockIdx.x * 128;
  const int n0   = blockIdx.y * 128;

  v8f acc[2][4] = {};

  for (int kk = 0; kk < DMODEL; kk += 32) {
    { // stage A: rows m0..m0+127, cols kk..kk+31 (fp32 -> bf16, packed)
      const int row  = tid >> 1;
      const int koff = (tid & 1) * 16;
      const f32x4* src =
          (const f32x4*)(X + (size_t)(m0 + row) * DMODEL + kk + koff);
      f32x4 fv[4];
#pragma unroll
      for (int j = 0; j < 4; ++j) fv[j] = src[j];
      u16x8 pk[2];
#pragma unroll
      for (int j = 0; j < 8; ++j) {
        pk[0][j] = f2bf(fv[j >> 2][j & 3]);
        pk[1][j] = f2bf(fv[2 + (j >> 2)][j & 3]);
      }
      *(u16x8*)&As[row][koff]     = pk[0];
      *(u16x8*)&As[row][koff + 8] = pk[1];
    }
    { // stage B^T: Bt[n][k] = W[kk+k][n0+n], k=0..31, n=0..127
      const int k  = tid >> 3;
      const int nf = (tid & 7) * 16;
      const f32x4* src =
          (const f32x4*)(W + (size_t)(kk + k) * DMODEL + n0 + nf);
      f32x4 fv[4];
#pragma unroll
      for (int j = 0; j < 4; ++j) fv[j] = src[j];
#pragma unroll
      for (int j = 0; j < 16; ++j) Bt[nf + j][k] = f2bf(fv[j >> 2][j & 3]);
    }
    __syncthreads();
    v16bf a[2];
#pragma unroll
    for (int r = 0; r < 2; ++r)
      a[r] = load_frag16(&As[wr * 32 + r * 16 + l15][lhi]);
#pragma unroll
    for (int g = 0; g < 4; ++g) {
      v16bf b = load_frag16(&Bt[wc * 64 + g * 16 + l15][lhi]);
#pragma unroll
      for (int r = 0; r < 2; ++r) acc[r][g] = wmma_bf16(a[r], b, acc[r][g]);
    }
    __syncthreads();
  }

  // epilogue: + bias, * scale, bf16 store (optionally transposed [B][D][T])
#pragma unroll
  for (int g = 0; g < 4; ++g) {
    const int n = n0 + wc * 64 + g * 16 + l15;
    const float bn = bias[n];
#pragma unroll
    for (int r = 0; r < 2; ++r) {
#pragma unroll
      for (int i = 0; i < 8; ++i) {
        const int m = m0 + wr * 32 + r * 16 + lhi + i;
        const unsigned short h = f2bf((acc[r][g][i] + bn) * scale);
        if (transpose_out) {
          const int bI = m >> 12;            // m / SEQ
          const int t  = m & (SEQ - 1);
          Out[((size_t)bI * DMODEL + n) * SEQ + t] = h;
        } else {
          Out[(size_t)m * DMODEL + n] = h;
        }
      }
    }
  }
}

// ---------------------------------------------------------------------------
// Pass 2: flash attention over bf16 Q/K (row-major) and V^T ([B][D][T]).
// Block: 16 query rows, 8 waves. Wave w owns d-slice [128w,128w+128) for both
// the QK^T inner dim (partials reduced via ds_add_f32) and the output cols.
// ---------------------------------------------------------------------------
__global__ __launch_bounds__(256)
void flash_attn_bf16(const unsigned short* __restrict__ Qb,
                     const unsigned short* __restrict__ Kb,
                     const unsigned short* __restrict__ Vt,
                     const int* __restrict__ mask,
                     float* __restrict__ Out) {
  __shared__ float Sls[16][32];                         // fp32 score tile
  __shared__ __align__(16) unsigned short Pls[16][40];  // bf16 probs (+pad)
  __shared__ float mrow[16], lrow[16], arow[16];

  const int tid  = threadIdx.x;
  const int wave = tid >> 5;
  const int lane = tid & 31;
  const int l15  = lane & 15;
  const int lhi  = (lane >> 4) * 8;
  const int q0   = blockIdx.x * 16;
  const int bI   = blockIdx.y;
  const int d0   = wave * 128;

  // Q fragments for this wave's d-slice live in registers (4 x v16bf).
  v16bf qfrag[4];
  {
    const unsigned short* qp =
        Qb + ((size_t)(bI * SEQ + q0 + l15)) * DMODEL + d0 + lhi;
#pragma unroll
    for (int f = 0; f < 4; ++f) qfrag[f] = load_frag16(qp + f * 32);
  }

  if (tid < 16) { mrow[tid] = -__builtin_inff(); lrow[tid] = 0.0f; }
  ((float*)Sls)[tid]       = 0.0f;  // 512 floats, 256 threads x 2
  ((float*)Sls)[tid + 256] = 0.0f;

  v8f acc[8] = {};

  for (int kb = 0; kb < SEQ; kb += 32) {
    // --- partial scores for this wave's d-slice (registers only) ---------
    v8f s[2] = {};
#pragma unroll
    for (int sdx = 0; sdx < 2; ++sdx) {
      const unsigned short* kp =
          Kb + ((size_t)(bI * SEQ + kb + sdx * 16 + l15)) * DMODEL + d0 + lhi;
#pragma unroll
      for (int f = 0; f < 4; ++f)
        s[sdx] = wmma_bf16(qfrag[f], load_frag16(kp + f * 32), s[sdx]);
    }
    if (kb + 32 < SEQ) { // prefetch next key block into caches
      __builtin_prefetch(
          Kb + ((size_t)(bI * SEQ + kb + 32 + l15)) * DMODEL + d0, 0, 1);
    }

    // [A] prior softmax finished zeroing Sls; prior PV finished with Pls
    __syncthreads();
#pragma unroll
    for (int sdx = 0; sdx < 2; ++sdx)
#pragma unroll
      for (int i = 0; i < 8; ++i)
        atomicAdd(&Sls[lhi + i][sdx * 16 + l15], s[sdx][i]);
    __syncthreads(); // [B] reduction complete

    // --- online softmax for the 16 query rows (vectorized LDS traffic) ---
    if (tid < 16) {
      const int r = tid;
      const int* mk = mask + bI * SEQ + kb;
      float sc[32];
      float mprev = mrow[r];
      float mnew  = mprev;
      const f32x4 zero4 = {};
#pragma unroll
      for (int j4 = 0; j4 < 8; ++j4) {
        f32x4 xv = *(const f32x4*)&Sls[r][j4 * 4];
        i32x4 mv = *(const i32x4*)(mk + j4 * 4);
#pragma unroll
        for (int e = 0; e < 4; ++e) {
          float x = (mv[e] == 0) ? -__builtin_inff() : xv[e];
          sc[j4 * 4 + e] = x;
          mnew = fmaxf(mnew, x);
        }
        *(f32x4*)&Sls[r][j4 * 4] = zero4; // re-zero for next reduction
      }
      const float msafe = (mnew == -__builtin_inff()) ? 0.0f : mnew;
      const float alpha =
          (mprev == -__builtin_inff()) ? 0.0f : __expf(mprev - msafe);
      float sum = 0.0f;
      u16x8 pk[4];
#pragma unroll
      for (int j = 0; j < 32; ++j) {
        const float p =
            (sc[j] == -__builtin_inff()) ? 0.0f : __expf(sc[j] - msafe);
        sum += p;
        pk[j >> 3][j & 7] = f2bf(p);
      }
#pragma unroll
      for (int g = 0; g < 4; ++g) *(u16x8*)&Pls[r][g * 8] = pk[g];
      lrow[r] = lrow[r] * alpha + sum;
      mrow[r] = mnew;
      arow[r] = alpha;
    }
    __syncthreads(); // [C] P and alpha ready

    // --- rescale accumulators and accumulate P @ V ----------------------
    float al[8];
#pragma unroll
    for (int i = 0; i < 8; ++i) al[i] = arow[lhi + i];
    const v16bf pfrag = load_frag16(&Pls[l15][lhi]);
#pragma unroll
    for (int g = 0; g < 8; ++g) {
#pragma unroll
      for (int i = 0; i < 8; ++i) acc[g][i] *= al[i];
      const unsigned short* vp =
          Vt + ((size_t)bI * DMODEL + d0 + g * 16 + l15) * SEQ + kb + lhi;
      acc[g] = wmma_bf16(pfrag, load_frag16(vp), acc[g]);
    }
  }

  // --- normalize and write fp32 output -----------------------------------
  float linv[8];
#pragma unroll
  for (int i = 0; i < 8; ++i) linv[i] = 1.0f / lrow[lhi + i];
#pragma unroll
  for (int g = 0; g < 8; ++g) {
    const int n = d0 + g * 16 + l15;
#pragma unroll
    for (int i = 0; i < 8; ++i) {
      Out[((size_t)(bI * SEQ + q0 + lhi + i)) * DMODEL + n] =
          acc[g][i] * linv[i];
    }
  }
}

// ---------------------------------------------------------------------------
extern "C" void kernel_launch(void* const* d_in, const int* in_sizes, int n_in,
                              void* d_out, int out_size, void* d_ws,
                              size_t ws_size, hipStream_t stream) {
  (void)in_sizes; (void)n_in; (void)out_size; (void)ws_size;

  const float* q    = (const float*)d_in[0];
  const float* k    = (const float*)d_in[1];
  const float* v    = (const float*)d_in[2];
  const float* Wq   = (const float*)d_in[3];
  const float* bq   = (const float*)d_in[4];
  const float* Wk   = (const float*)d_in[5];
  const float* bk   = (const float*)d_in[6];
  const float* Wv   = (const float*)d_in[7];
  const float* bv   = (const float*)d_in[8];
  const int*   mask = (const int*)d_in[9];
  float*       out  = (float*)d_out;

  const size_t nTok = (size_t)NBATCH * SEQ;           // 16384
  unsigned short* Qb = (unsigned short*)d_ws;          // [B*T, D] bf16
  unsigned short* Kb = Qb + nTok * DMODEL;             // [B*T, D] bf16
  unsigned short* Vt = Kb + nTok * DMODEL;             // [B, D, T] bf16

  const dim3 pgrid(nTok / 128, DMODEL / 128);          // (128, 8)
  const float qscale = 0.03125f;                       // 1/sqrt(1024)

  proj_gemm_bf16<<<pgrid, 256, 0, stream>>>(q, Wq, bq, Qb, qscale, 0);
  proj_gemm_bf16<<<pgrid, 256, 0, stream>>>(k, Wk, bk, Kb, 1.0f, 0);
  proj_gemm_bf16<<<pgrid, 256, 0, stream>>>(v, Wv, bv, Vt, 1.0f, 1);

  flash_attn_bf16<<<dim3(SEQ / 16, NBATCH), 256, 0, stream>>>(
      Qb, Kb, Vt, mask, out);
}
